// Block_7215545057583
// MI455X (gfx1250) — matrix operand order repaired and checked by reading, not compiled, not run on hardware
//
#include <hip/hip_runtime.h>
#include <hip/hip_bf16.h>

// ---------------------------------------------------------------------------
// Types for CDNA5 WMMA (wave32): 16x16x32 bf16 -> f32
// ---------------------------------------------------------------------------
typedef __attribute__((ext_vector_type(16))) __bf16 v16bf;
typedef __attribute__((ext_vector_type(8)))  __bf16 bf16x8;
typedef __attribute__((ext_vector_type(4)))  __bf16 bf16x4;
typedef __attribute__((ext_vector_type(8)))  float  v8f;
typedef __attribute__((ext_vector_type(4)))  unsigned u32x4;
typedef __attribute__((ext_vector_type(4)))  int      i32x4;
typedef __attribute__((ext_vector_type(8)))  int      i32x8;

union AFrag { v16bf v; bf16x8 h[2]; };

static __device__ __forceinline__ __bf16 f2bf(float f) {
  unsigned u = __builtin_bit_cast(unsigned, f);
  unsigned r = (u + 0x7FFFu + ((u >> 16) & 1u)) >> 16;
  unsigned short s = (unsigned short)r;
  return __builtin_bit_cast(__bf16, s);
}

static __device__ __forceinline__ v8f wmma_bf16(v16bf a, v16bf b, v8f c) {
  return __builtin_amdgcn_wmma_f32_16x16x32_bf16(
      /*neg_a=*/false, a, /*neg_b=*/false, b,
      /*c_mod=*/(short)0, c, /*reuse_a=*/false, /*reuse_b=*/false);
}

// generic LDS pointer -> 32-bit LDS byte offset (flat LDS aperture keeps the
// workgroup-relative offset in addr[31:0] per CDNA5 flat addressing rules)
static __device__ __forceinline__ unsigned ldsoff(const void* p) {
  return (unsigned)(unsigned long long)p;
}

// ---------------------------------------------------------------------------
// Tensor Data Mover: async DMA of a 2-D bf16 tile (rows x cols, row stride in
// elements) from global memory into LDS.  Descriptor per CDNA5 ISA ch.8:
//   group0: count=1 | lds_addr | global_addr | type=2
//   group1: data_size=2B, tensor_dim0/1, tile_dim0/1, tensor_dim0_stride
//   groups 2/3: zero (2-D tensor, VADDR2/3 semantics handled by builtin)
// Tracked with TENSORcnt; pipelined with s_wait_tensorcnt.
// ---------------------------------------------------------------------------
static __device__ __forceinline__ void tdm_load_2d_bf16(
    unsigned lds_addr, const void* gptr,
    unsigned tile_d0 /*cols, elems*/, unsigned tile_d1 /*rows*/,
    unsigned long long stride0 /*elems between rows*/) {
  unsigned long long ga = (unsigned long long)gptr;
  u32x4 g0;
  g0[0] = 1u;                                             // count=1, user mode
  g0[1] = lds_addr;                                       // lds_addr (bytes)
  g0[2] = (unsigned)(ga & 0xFFFFFFFFu);                   // global_addr[31:0]
  g0[3] = (unsigned)((ga >> 32) & 0x01FFFFFFu) | (2u << 30);  // [56:32] | type=2
  i32x8 g1;
  g1[0] = (int)(1u << 16);                                // wg_mask=0, data_size=1 (2B)
  g1[1] = (int)((tile_d0 & 0xFFFFu) << 16);               // tensor_dim0[15:0]
  g1[2] = (int)(((tile_d0 >> 16) & 0xFFFFu) | ((tile_d1 & 0xFFFFu) << 16));
  g1[3] = (int)(((tile_d1 >> 16) & 0xFFFFu) | ((tile_d0 & 0xFFFFu) << 16)); // tile_dim0
  g1[4] = (int)(tile_d1 & 0xFFFFu);                       // tile_dim1, tile_dim2=0
  g1[5] = (int)(unsigned)(stride0 & 0xFFFFFFFFull);       // dim0 stride [31:0]
  g1[6] = (int)(unsigned)((stride0 >> 32) & 0xFFFFull);   // stride [47:32]
  g1[7] = 0;
  i32x4 z4 = {0, 0, 0, 0};
#if __clang_major__ >= 23
  i32x8 z8 = {0, 0, 0, 0, 0, 0, 0, 0};
  __builtin_amdgcn_tensor_load_to_lds(g0, g1, z4, z4, z8, 0);
#else
  __builtin_amdgcn_tensor_load_to_lds(g0, g1, z4, z4, 0);
#endif
}

// ---------------------------------------------------------------------------
// Transposing f32 -> bf16 weight conversion: out[N][K] = (bf16)in[K][N]
// ---------------------------------------------------------------------------
__global__ __launch_bounds__(256)
void transpose_f32_to_bf16_kernel(const float* __restrict__ in, __bf16* __restrict__ out,
                                  int K, int N) {
  __shared__ __bf16 tile[64][68];
  int k0 = blockIdx.y * 64, n0 = blockIdx.x * 64;
  int tr = threadIdx.x >> 4;          // 0..15
  int tc = (threadIdx.x & 15) * 4;    // 0..60
#pragma unroll
  for (int rr = 0; rr < 64; rr += 16) {
    float4 vv = *(const float4*)(in + (size_t)(k0 + tr + rr) * N + n0 + tc);
    tile[tr + rr][tc + 0] = f2bf(vv.x);
    tile[tr + rr][tc + 1] = f2bf(vv.y);
    tile[tr + rr][tc + 2] = f2bf(vv.z);
    tile[tr + rr][tc + 3] = f2bf(vv.w);
  }
  __syncthreads();
#pragma unroll
  for (int rr = 0; rr < 64; rr += 16) {
    int n = tr + rr;
    bf16x4 o;
    o[0] = tile[tc + 0][n];
    o[1] = tile[tc + 1][n];
    o[2] = tile[tc + 2][n];
    o[3] = tile[tc + 3][n];
    *(bf16x4*)(out + (size_t)(n0 + n) * K + k0 + tc) = o;
  }
}

// ---------------------------------------------------------------------------
// LayerNorm over C=1024, one block (8 waves) per row, bf16 output
// ---------------------------------------------------------------------------
__global__ __launch_bounds__(256)
void layernorm_bf16_kernel(const float* __restrict__ x, const float* __restrict__ g,
                           const float* __restrict__ b, __bf16* __restrict__ out) {
  const int C = 1024;
  int row = blockIdx.x;
  int t = threadIdx.x;
  const float* xr = x + (size_t)row * C;
  float4 xv = ((const float4*)xr)[t];
  float s  = xv.x + xv.y + xv.z + xv.w;
  float s2 = xv.x*xv.x + xv.y*xv.y + xv.z*xv.z + xv.w*xv.w;
#pragma unroll
  for (int m = 1; m < 32; m <<= 1) { s += __shfl_xor(s, m); s2 += __shfl_xor(s2, m); }
  __shared__ float rs_[8], rs2_[8];
  int wid = t >> 5, lane = t & 31;
  if (lane == 0) { rs_[wid] = s; rs2_[wid] = s2; }
  __syncthreads();
  float ts = 0.f, ts2 = 0.f;
#pragma unroll
  for (int i = 0; i < 8; ++i) { ts += rs_[i]; ts2 += rs2_[i]; }
  float mu  = ts * (1.0f / C);
  float var = ts2 * (1.0f / C) - mu * mu;
  float inv = rsqrtf(var + 1e-5f);
  float4 gv = ((const float4*)g)[t];
  float4 bv = ((const float4*)b)[t];
  bf16x4 o;
  o[0] = f2bf((xv.x - mu) * inv * gv.x + bv.x);
  o[1] = f2bf((xv.y - mu) * inv * gv.y + bv.y);
  o[2] = f2bf((xv.z - mu) * inv * gv.z + bv.z);
  o[3] = f2bf((xv.w - mu) * inv * gv.w + bv.w);
  *(bf16x4*)(out + (size_t)row * C + t * 4) = o;
}

// ---------------------------------------------------------------------------
// WMMA GEMM: C[M,N] = A[M,K](bf16) @ WT[N,K](bf16, pre-transposed) + bias.
// BM=128 BN=128 BK=32, 256 threads = 8 waves (4 M-slots x 2 N-slots),
// each wave computes 32x64 = 2x4 WMMA tiles.  Both input tiles are staged
// into LDS by the Tensor Data Mover, double-buffered on TENSORcnt.
// ---------------------------------------------------------------------------
enum { EPI_BF16 = 0, EPI_RES_F32 = 1, EPI_GELU_BF16 = 2, EPI_BF16_VT = 3 };

template <int EPI>
__global__ __launch_bounds__(256)
void gemm_bf16_kernel(const __bf16* __restrict__ A, const __bf16* __restrict__ WT,
                      const float* __restrict__ bias, const float* __restrict__ res,
                      void* __restrict__ outv, int M, int N, int K) {
  __shared__ __align__(32) __bf16 As[2][128][32];   // As[buf][m][k]
  __shared__ __align__(32) __bf16 Bs[2][128][32];   // Bs[buf][n][k]
  int tid = threadIdx.x;
  int wid = tid >> 5, lane = tid & 31;
  int hl = lane >> 4, l16 = lane & 15;
  int m0 = blockIdx.y * 128, n0 = blockIdx.x * 128;
  int wm = (wid & 3) * 32;
  int wn = (wid >> 2) * 64;

  v8f acc[2][4] = {};
  int steps = K >> 5;

  auto issue = [&](int i) {
    int buf = i & 1;
    tdm_load_2d_bf16(ldsoff(&As[buf][0][0]),
                     A + (size_t)m0 * K + (size_t)i * 32, 32, 128, (unsigned long long)K);
    tdm_load_2d_bf16(ldsoff(&Bs[buf][0][0]),
                     WT + (size_t)n0 * K + (size_t)i * 32, 32, 128, (unsigned long long)K);
  };

  if (tid < 32) issue(0);   // wave 0 drives the TDM

  for (int i = 0; i < steps; ++i) {
    int buf = i & 1;
    if (tid < 32) {
      if (i + 1 < steps) {
        issue(i + 1);                               // into the other buffer
        __builtin_amdgcn_s_wait_tensorcnt((short)2);  // tile i (oldest 2) done
      } else {
        __builtin_amdgcn_s_wait_tensorcnt((short)0);
      }
    }
    __syncthreads();   // publish DMA'd tile to all waves

    AFrag af[2];
#pragma unroll
    for (int mt = 0; mt < 2; ++mt) {
      int m = wm + mt * 16 + l16;
      af[mt].h[0] = *(const bf16x8*)&As[buf][m][hl * 8];
      af[mt].h[1] = *(const bf16x8*)&As[buf][m][16 + hl * 8];
    }
#pragma unroll
    for (int nt = 0; nt < 4; ++nt) {
      int n = wn + nt * 16 + l16;
      AFrag bfr;
      bfr.h[0] = *(const bf16x8*)&Bs[buf][n][hl * 8];
      bfr.h[1] = *(const bf16x8*)&Bs[buf][n][16 + hl * 8];
      v16bf bfv = bfr.v;
#pragma unroll
      for (int mt = 0; mt < 2; ++mt)
        acc[mt][nt] = wmma_bf16(af[mt].v, bfv, acc[mt][nt]);
    }
    __syncthreads();   // all waves done reading buf before it is DMA'd again
  }

  // Epilogue. C/D layout: VGPR r, lanes0-15 -> row r, lanes16-31 -> row r+8.
#pragma unroll
  for (int mt = 0; mt < 2; ++mt) {
#pragma unroll
    for (int nt = 0; nt < 4; ++nt) {
#pragma unroll
      for (int r = 0; r < 8; ++r) {
        int m = m0 + wm + mt * 16 + r + hl * 8;
        int n = n0 + wn + nt * 16 + l16;
        size_t idx = (size_t)m * N + n;
        float vv = acc[mt][nt][r] + bias[n];
        if (EPI == EPI_RES_F32) {
          ((float*)outv)[idx] = vv + res[idx];
        } else if (EPI == EPI_GELU_BF16) {
          float gl = 0.5f * vv * (1.0f + erff(vv * 0.70710678f));
          ((__bf16*)outv)[idx] = f2bf(gl);
        } else if (EPI == EPI_BF16_VT) {
          // transposed store for V: out[B*H][64][T], T=2048
          size_t vidx = (((size_t)(m >> 11) * 16 + (size_t)(n >> 6)) * 64 + (n & 63)) * 2048
                        + (m & 2047);
          ((__bf16*)outv)[vidx] = f2bf(vv);
        } else {
          ((__bf16*)outv)[idx] = f2bf(vv);
        }
      }
    }
  }
}

// ---------------------------------------------------------------------------
// Flash attention (causal).  q,k: bf16 [B*T, 1024] (col = h*64+d);
// vT: bf16 [B*H][64][T] (pre-transposed by the V-projection epilogue).
// Grid: (T/128, B*H). WG = 128 query rows of one (b,h); wave owns 16 rows.
// K and V tiles are DMA'd by the TDM, double-buffered on TENSORcnt.
// S = Q K^T via WMMA, online softmax with shfl row reductions, P re-laid-out
// through per-wave LDS, O += P@V via WMMA.
// ---------------------------------------------------------------------------
__global__ __launch_bounds__(256)
void flash_attn_kernel(const __bf16* __restrict__ q, const __bf16* __restrict__ k,
                       const __bf16* __restrict__ vT, __bf16* __restrict__ y) {
  const int T = 2048, C = 1024;
  int bh = blockIdx.y;
  int bb = bh >> 4, hh = bh & 15;
  int q0 = blockIdx.x * 128;
  int tid = threadIdx.x, wid = tid >> 5, lane = tid & 31;
  int hl = lane >> 4, l16 = lane & 15;

  __shared__ __align__(32) __bf16 Kt[2][64][64];   // Kt[buf][key][d]
  __shared__ __align__(32) __bf16 Vt[2][64][64];   // Vt[buf][d][key]
  __shared__ __align__(32) __bf16 Sw[8][16][64];   // per-wave P staging

  // Q fragments for this wave's 16 rows (A layout: m = l16, k split by half)
  AFrag qf[2];
  {
    const __bf16* qb = q + ((size_t)(bb * T) + q0 + wid * 16 + l16) * C + hh * 64;
#pragma unroll
    for (int st = 0; st < 2; ++st) {
      qf[st].h[0] = *(const bf16x8*)(qb + st * 32 + hl * 8);
      qf[st].h[1] = *(const bf16x8*)(qb + st * 32 + 16 + hl * 8);
    }
  }

  float m_run[8], l_run[8];
#pragma unroll
  for (int r = 0; r < 8; ++r) { m_run[r] = -1e30f; l_run[r] = 0.f; }
  v8f o[4] = {};

  int nsteps = (q0 >> 6) + 2;   // key blocks of 64, up to q0+127

  auto issueKV = [&](int i) {
    int buf = i & 1;
    int ks = i * 64;
    tdm_load_2d_bf16(ldsoff(&Kt[buf][0][0]),
                     k + ((size_t)(bb * T) + ks) * C + hh * 64, 64, 64, (unsigned long long)C);
    tdm_load_2d_bf16(ldsoff(&Vt[buf][0][0]),
                     vT + ((size_t)bh * 64) * T + ks, 64, 64, (unsigned long long)T);
  };

  if (tid < 32) issueKV(0);

  for (int i = 0; i < nsteps; ++i) {
    int buf = i & 1;
    int ks = i * 64;
    if (tid < 32) {
      if (i + 1 < nsteps) {
        issueKV(i + 1);
        __builtin_amdgcn_s_wait_tensorcnt((short)2);
      } else {
        __builtin_amdgcn_s_wait_tensorcnt((short)0);
      }
    }
    __syncthreads();

    // S = Q K^T : 4 key-tiles x 2 k-steps
    v8f s[4] = {};
#pragma unroll
    for (int nt = 0; nt < 4; ++nt) {
#pragma unroll
      for (int st = 0; st < 2; ++st) {
        AFrag kfr;
        kfr.h[0] = *(const bf16x8*)&Kt[buf][nt * 16 + l16][st * 32 + hl * 8];
        kfr.h[1] = *(const bf16x8*)&Kt[buf][nt * 16 + l16][st * 32 + 16 + hl * 8];
        v16bf kf = kfr.v;
        s[nt] = wmma_bf16(qf[st].v, kf, s[nt]);
      }
    }

    // scale, causal mask, online softmax (rows live in 16-lane halves)
#pragma unroll
    for (int r = 0; r < 8; ++r) {
      int qrow = q0 + wid * 16 + r + hl * 8;
      float rmax = m_run[r];
#pragma unroll
      for (int nt = 0; nt < 4; ++nt) {
        int key = ks + nt * 16 + l16;
        float sv = s[nt][r] * 0.125f;           // 1/sqrt(64)
        sv = (key <= qrow) ? sv : -1e30f;
        s[nt][r] = sv;
        rmax = fmaxf(rmax, sv);
      }
      rmax = fmaxf(rmax, __shfl_xor(rmax, 1));
      rmax = fmaxf(rmax, __shfl_xor(rmax, 2));
      rmax = fmaxf(rmax, __shfl_xor(rmax, 4));
      rmax = fmaxf(rmax, __shfl_xor(rmax, 8));
      float alpha = __expf(m_run[r] - rmax);
      float rsum = 0.f;
#pragma unroll
      for (int nt = 0; nt < 4; ++nt) {
        float p = __expf(s[nt][r] - rmax);
        s[nt][r] = p;
        rsum += p;
      }
      rsum += __shfl_xor(rsum, 1);
      rsum += __shfl_xor(rsum, 2);
      rsum += __shfl_xor(rsum, 4);
      rsum += __shfl_xor(rsum, 8);
      l_run[r] = l_run[r] * alpha + rsum;
      m_run[r] = rmax;
#pragma unroll
      for (int dt = 0; dt < 4; ++dt) o[dt][r] *= alpha;   // rescale O rows
    }

    // P -> per-wave LDS (C/D layout -> memory), then reload as A fragments
#pragma unroll
    for (int nt = 0; nt < 4; ++nt)
#pragma unroll
      for (int r = 0; r < 8; ++r)
        Sw[wid][r + hl * 8][nt * 16 + l16] = f2bf(s[nt][r]);

    AFrag pf[2];
#pragma unroll
    for (int st = 0; st < 2; ++st) {
      pf[st].h[0] = *(const bf16x8*)&Sw[wid][l16][st * 32 + hl * 8];
      pf[st].h[1] = *(const bf16x8*)&Sw[wid][l16][st * 32 + 16 + hl * 8];
    }

    // O += P @ V  (K-dim = 64 keys, N = d; Vt[d][key] makes B-frags contiguous)
#pragma unroll
    for (int dt = 0; dt < 4; ++dt) {
#pragma unroll
      for (int st = 0; st < 2; ++st) {
        AFrag vfr;
        vfr.h[0] = *(const bf16x8*)&Vt[buf][dt * 16 + l16][st * 32 + hl * 8];
        vfr.h[1] = *(const bf16x8*)&Vt[buf][dt * 16 + l16][st * 32 + 16 + hl * 8];
        v16bf vf = vfr.v;
        o[dt] = wmma_bf16(pf[st].v, vf, o[dt]);
      }
    }
    __syncthreads();   // done reading buf before it is DMA'd again
  }

  // epilogue: O / l, write y[B*T, C] at col h*64+d
#pragma unroll
  for (int dt = 0; dt < 4; ++dt) {
#pragma unroll
    for (int r = 0; r < 8; ++r) {
      int qrow = q0 + wid * 16 + r + hl * 8;
      int d = dt * 16 + l16;
      y[((size_t)(bb * T) + qrow) * C + hh * 64 + d] = f2bf(o[dt][r] / l_run[r]);
    }
  }
}

// ---------------------------------------------------------------------------
// Host orchestration
// ---------------------------------------------------------------------------
extern "C" void kernel_launch(void* const* d_in, const int* in_sizes, int n_in,
                              void* d_out, int out_size, void* d_ws, size_t ws_size,
                              hipStream_t stream) {
  (void)in_sizes; (void)n_in; (void)out_size; (void)ws_size;
  const int B = 2, T = 2048, C = 1024, DFF = 4096;
  const int M = B * T;

  const float* x    = (const float*)d_in[0];
  const float* ln1g = (const float*)d_in[1];
  const float* ln1b = (const float*)d_in[2];
  const float* Wq   = (const float*)d_in[3];
  const float* bq   = (const float*)d_in[4];
  const float* Wk   = (const float*)d_in[5];
  const float* bk   = (const float*)d_in[6];
  const float* Wv   = (const float*)d_in[7];
  const float* bv   = (const float*)d_in[8];
  const float* Wo   = (const float*)d_in[9];
  const float* bo   = (const float*)d_in[10];
  const float* ln2g = (const float*)d_in[11];
  const float* ln2b = (const float*)d_in[12];
  const float* W1   = (const float*)d_in[13];
  const float* b1   = (const float*)d_in[14];
  const float* W2   = (const float*)d_in[15];
  const float* b2   = (const float*)d_in[16];

  char* ws = (char*)d_ws;
  const size_t MB = 1ull << 20;
  __bf16* wq_bf = (__bf16*)(ws + 0 * MB);    // all weights pre-transposed [N][K]
  __bf16* wk_bf = (__bf16*)(ws + 2 * MB);
  __bf16* wv_bf = (__bf16*)(ws + 4 * MB);
  __bf16* wo_bf = (__bf16*)(ws + 6 * MB);
  __bf16* w1_bf = (__bf16*)(ws + 8 * MB);
  __bf16* w2_bf = (__bf16*)(ws + 16 * MB);
  __bf16* xln   = (__bf16*)(ws + 24 * MB);   // ln1 out; reused as ln2 out (h)
  __bf16* qb    = (__bf16*)(ws + 32 * MB);
  __bf16* kb    = (__bf16*)(ws + 40 * MB);
  __bf16* vTb   = (__bf16*)(ws + 48 * MB);   // V, [B*H][64][T] layout
  __bf16* yb    = (__bf16*)(ws + 56 * MB);
  float*  x2    = (float*) (ws + 64 * MB);   // 16 MB fp32 residual stream
  __bf16* hb    = xln;                       // ln2 output (8 MB)
  __bf16* h2b   = qb;                        // gelu output (32 MB, reuses q/k/vT/y)

  auto tp = [&](const float* src, __bf16* dst, int K_, int N_) {
    transpose_f32_to_bf16_kernel<<<dim3(N_ / 64, K_ / 64), 256, 0, stream>>>(src, dst, K_, N_);
  };
  tp(Wq, wq_bf, C, C);
  tp(Wk, wk_bf, C, C);
  tp(Wv, wv_bf, C, C);
  tp(Wo, wo_bf, C, C);
  tp(W1, w1_bf, C, DFF);
  tp(W2, w2_bf, DFF, C);

  // ln1
  layernorm_bf16_kernel<<<M, 256, 0, stream>>>(x, ln1g, ln1b, xln);

  // q, k, v projections (v written transposed for the attention V tiles)
  dim3 gC(C / 128, M / 128);
  gemm_bf16_kernel<EPI_BF16>   <<<gC, 256, 0, stream>>>(xln, wq_bf, bq, nullptr, (void*)qb,  M, C, C);
  gemm_bf16_kernel<EPI_BF16>   <<<gC, 256, 0, stream>>>(xln, wk_bf, bk, nullptr, (void*)kb,  M, C, C);
  gemm_bf16_kernel<EPI_BF16_VT><<<gC, 256, 0, stream>>>(xln, wv_bf, bv, nullptr, (void*)vTb, M, C, C);

  // attention
  flash_attn_kernel<<<dim3(T / 128, B * 16), 256, 0, stream>>>(qb, kb, vTb, yb);

  // out-proj + residual -> x2 (fp32)
  gemm_bf16_kernel<EPI_RES_F32><<<gC, 256, 0, stream>>>(yb, wo_bf, bo, x, (void*)x2, M, C, C);

  // ln2
  layernorm_bf16_kernel<<<M, 256, 0, stream>>>(x2, ln2g, ln2b, hb);

  // MLP up + GELU(exact)
  dim3 gF(DFF / 128, M / 128);
  gemm_bf16_kernel<EPI_GELU_BF16><<<gF, 256, 0, stream>>>(hb, w1_bf, b1, nullptr, (void*)h2b, M, DFF, C);

  // MLP down + residual -> output (fp32)
  gemm_bf16_kernel<EPI_RES_F32><<<gC, 256, 0, stream>>>(h2b, w2_bf, b2, x2, d_out, M, C, DFF);
}
